// Audio2Motion_RetNet_37924561223718
// MI455X (gfx1250) — matrix operand order repaired and checked by reading, not compiled, not run on hardware
//
#include <hip/hip_runtime.h>
#include <hip/hip_bf16.h>
#include <cstdint>

// ---------------------------------------------------------------------------
// Audio2Motion RetNet forward — CDNA5 (gfx1250) WMMA implementation.
// GEMMs: v_wmma_f32_16x16x32_f16 (f16 in, f32 accum), A-panel staged into LDS
// by the Tensor Data Mover (tensor_load_to_lds + s_wait_tensorcnt), B-panel
// staged transposed by vector loads. Retention fused with WMMA.
// ---------------------------------------------------------------------------

typedef _Float16 half_t;
typedef __attribute__((ext_vector_type(16))) _Float16 v16h;
typedef __attribute__((ext_vector_type(8)))  _Float16 v8h;
typedef __attribute__((ext_vector_type(4)))  _Float16 v4h;
typedef __attribute__((ext_vector_type(8)))  float    v8f;
typedef __attribute__((ext_vector_type(4)))  unsigned int u32x4;
typedef __attribute__((ext_vector_type(8)))  int i32x8;
typedef __attribute__((ext_vector_type(4)))  int i32x4;

union V16U { v16h v; v8h h[2]; };

#define DEV __device__ __forceinline__

#if defined(__has_builtin)
#if __has_builtin(__builtin_amdgcn_tensor_load_to_lds)
#define USE_TDM 1
#else
#define USE_TDM 0
#endif
#else
#define USE_TDM 0
#endif

DEV v8f wmma_f16(v16h a, v16h b, v8f c) {
  // (neg_a, A, neg_b, B, c_mod, C, reuse_a, reuse_b)
  return __builtin_amdgcn_wmma_f32_16x16x32_f16(false, a, false, b, (short)0, c,
                                                false, false);
}

template <int ACT> DEV float activate(float v) {
  if (ACT == 1) return v >= 0.f ? v : 0.01f * v;                       // LeakyReLU
  if (ACT == 2) {                                                      // gelu (tanh)
    float t = 0.7978845608028654f * (v + 0.044715f * v * v * v);
    return 0.5f * v * (1.f + tanhf(t));
  }
  if (ACT == 3) return v / (1.f + expf(-v));                           // silu
  return v;
}

#if USE_TDM
// TDM: DMA a 128-row x 32-half (64B) tile, row stride = lda halves, from
// global into LDS. LDS pad: 4 DWORDs after every 16 DWORDs -> 40-half rows,
// matching the WMMA fragment-load layout. D# packing per ISA 08 §8.
DEV void tdm_load_tileA(const half_t* gsrc, unsigned int ldsOff, int lda) {
  uint64_t ga = (uint64_t)(uintptr_t)gsrc;
  u32x4 g0;
  g0.x = 1u;                                   // count=1, user descriptor
  g0.y = ldsOff;                               // lds_addr [63:32]
  g0.z = (unsigned int)ga;                     // global_addr [95:64]
  g0.w = (unsigned int)(ga >> 32) | (2u << 30);  // addr hi + type=2 ("image")
  unsigned int td0 = (unsigned int)lda;        // tensor_dim0 (elements/row)
  unsigned int td1 = 1u << 20;                 // tensor_dim1 (rows, large)
  i32x8 g1;
  g1[0] = (int)((1u << 16)      // data_size = 2 bytes
                | (1u << 20)    // pad_enable
                | (3u << 22)    // pad_interval: 16 DWORDs
                | (3u << 25));  // pad_amount: 4 DWORDs
  g1[1] = (int)((td0 & 0xFFFFu) << 16);
  g1[2] = (int)((td0 >> 16) | ((td1 & 0xFFFFu) << 16));
  g1[3] = (int)((td1 >> 16) | (32u << 16));    // tile_dim0 = 32 elements
  g1[4] = 128;                                 // tile_dim1 = 128 rows
  g1[5] = (int)td0;                            // tensor_dim0_stride lo
  g1[6] = 0;
  g1[7] = 0;
  i32x4 gz = {0, 0, 0, 0};
#if defined(__clang_major__) && (__clang_major__ >= 23)
  i32x8 gz8 = {0, 0, 0, 0, 0, 0, 0, 0};
  __builtin_amdgcn_tensor_load_to_lds(g0, g1, gz, gz, gz8, 0);
#else
  __builtin_amdgcn_tensor_load_to_lds(g0, g1, gz, gz, 0);
#endif
}
#endif

// ---------------------------------------------------------------------------
// Generic WMMA GEMM:  C[M,N] = act(A[M,K] * W[K,N] + bias) (+ residual)
// Block tile 128x64, 8 waves, each wave 32x32 (2x2 wmma tiles), K step 32.
// Double-buffered LDS; padded strides (40 halves) to dodge bank conflicts.
// A staged by the TDM (wave 0 issues, TENSORcnt overlapped with compute).
// ---------------------------------------------------------------------------
template <int ACT, bool OUT16, bool RES>
__global__ __launch_bounds__(256) void gemm_wmma(
    const half_t* __restrict__ A, int lda, const half_t* __restrict__ W, int ldb,
    const float* __restrict__ bias, const float* __restrict__ resid, int ldr,
    void* __restrict__ Cout, int ldc, int K) {
  __shared__ half_t As[2][128 * 40];
  __shared__ half_t Bs[2][64 * 40];

  const int tid = threadIdx.x;
  const int lane = tid & 31, wave = tid >> 5;
  const int wm = wave >> 1, wn = wave & 1;
  const int m0 = blockIdx.y * 128, n0 = blockIdx.x * 64;
  const bool tdmWave = (wave == 0);
  (void)tdmWave;

#if !USE_TDM
  auto loadA = [&](int kt, int buf) {
#pragma unroll
    for (int i = 0; i < 2; i++) {
      int c = tid + i * 256;                 // 512 chunks of 8 halves
      int row = c >> 2, c8 = (c & 3) << 3;
      v8h val = *(const v8h*)(A + (size_t)(m0 + row) * lda + kt + c8);
      *(v8h*)&As[buf][row * 40 + c8] = val;
    }
  };
#endif
  auto loadB = [&](int kt, int buf) {        // store transposed: Bs[n][k]
#pragma unroll
    for (int i = 0; i < 2; i++) {
      int c = tid + i * 256;                 // 512 chunks of 4 halves
      int k = c >> 4, n4 = (c & 15) << 2;
      v4h val = *(const v4h*)(W + (size_t)(kt + k) * ldb + n0 + n4);
#pragma unroll
      for (int j = 0; j < 4; j++) Bs[buf][(n4 + j) * 40 + k] = val[j];
    }
  };

  const v8f vzero = {0.f, 0.f, 0.f, 0.f, 0.f, 0.f, 0.f, 0.f};
  v8f acc[2][2];
#pragma unroll
  for (int i = 0; i < 2; i++)
#pragma unroll
    for (int j = 0; j < 2; j++) acc[i][j] = vzero;

#if USE_TDM
  unsigned int ldsA0 = (unsigned int)(uintptr_t)&As[0][0];
  unsigned int ldsA1 = (unsigned int)(uintptr_t)&As[1][0];
  if (tdmWave) tdm_load_tileA(A + (size_t)m0 * lda, ldsA0, lda);
#else
  loadA(0, 0);
#endif
  loadB(0, 0);

  for (int kt = 0; kt < K; kt += 32) {
    int buf = (kt >> 5) & 1;
#if USE_TDM
    if (tdmWave) __builtin_amdgcn_s_wait_tensorcnt((short)0);
#endif
    __syncthreads();
    if (kt + 32 < K) {
#if USE_TDM
      if (tdmWave)
        tdm_load_tileA(A + (size_t)m0 * lda + (kt + 32), buf ? ldsA0 : ldsA1, lda);
#else
      loadA(kt + 32, buf ^ 1);
#endif
      loadB(kt + 32, buf ^ 1);
    }
    if (kt + 64 < K)  // hint next stage into cache (global_prefetch_b8)
      __builtin_prefetch(A + (size_t)(m0 + (tid >> 2)) * lda + kt + 64, 0, 0);

#pragma unroll
    for (int sm = 0; sm < 2; sm++) {
      const half_t* arow =
          &As[buf][(wm * 32 + sm * 16 + (lane & 15)) * 40 + ((lane < 16) ? 0 : 8)];
      V16U af;
      af.h[0] = *(const v8h*)arow;
      af.h[1] = *(const v8h*)(arow + 16);
#pragma unroll
      for (int sn = 0; sn < 2; sn++) {
        const half_t* brow =
            &Bs[buf][(wn * 32 + sn * 16 + (lane & 15)) * 40 + ((lane < 16) ? 0 : 8)];
        V16U bf;
        bf.h[0] = *(const v8h*)brow;
        bf.h[1] = *(const v8h*)(brow + 16);
        acc[sm][sn] = wmma_f16(af.v, bf.v, acc[sm][sn]);
      }
    }
  }

  // Epilogue: C fragment lanes 0-15 -> M=r, 16-31 -> M=8+r; N = lane%16.
#pragma unroll
  for (int sm = 0; sm < 2; sm++)
#pragma unroll
    for (int sn = 0; sn < 2; sn++) {
      v8f cacc = acc[sm][sn];
      int col = n0 + wn * 32 + sn * 16 + (lane & 15);
      float bb = bias ? bias[col] : 0.f;
      int rbase = m0 + wm * 32 + sm * 16 + ((lane >> 4) << 3);
#pragma unroll
      for (int r = 0; r < 8; r++) {
        float v = activate<ACT>(cacc[r] + bb);
        if (RES) v += resid[(size_t)(rbase + r) * ldr + col];
        if (OUT16)
          ((half_t*)Cout)[(size_t)(rbase + r) * ldc + col] = (half_t)v;
        else
          ((float*)Cout)[(size_t)(rbase + r) * ldc + col] = v;
      }
    }
}

// ---------------------------------------------------------------------------
// Retention: per (b,h) causal decayed attention. ret = (QK^T/8 ∘ D) V.
// Block = 4 waves, each wave one 16-row query tile; 64-key chunks in LDS.
// ---------------------------------------------------------------------------
__global__ __launch_bounds__(128) void retention_kernel(
    const half_t* __restrict__ Qg, const half_t* __restrict__ Kg,
    const half_t* __restrict__ Vg, float* __restrict__ ret) {
  const int T = 1024;
  __shared__ half_t Ks[64 * 72];   // [key][d]
  __shared__ half_t Vs[64 * 72];   // [d][key] (transposed)
  __shared__ half_t Ss[4][16 * 72];

  int bh = blockIdx.y;
  int b = bh >> 3, h = bh & 7;
  int q0blk = blockIdx.x * 64;
  int tid = threadIdx.x, lane = tid & 31, wave = tid >> 5;
  int q0 = q0blk + wave * 16;
  float gamma = 1.f - exp2f(-5.f - (float)h);
  float l2g = log2f(gamma);

  // Preload Q A-fragments (16 rows x 64 features = two k-stages).
  V16U qf[2];
  {
    const half_t* qrow = Qg + ((size_t)(b * T + q0 + (lane & 15))) * 512 + h * 64 +
                         ((lane < 16) ? 0 : 8);
#pragma unroll
    for (int st = 0; st < 2; st++) {
      qf[st].h[0] = *(const v8h*)(qrow + st * 32);
      qf[st].h[1] = *(const v8h*)(qrow + st * 32 + 16);
    }
  }

  const v8f vzero = {0.f, 0.f, 0.f, 0.f, 0.f, 0.f, 0.f, 0.f};
  v8f racc[4];
#pragma unroll
  for (int i = 0; i < 4; i++) racc[i] = vzero;

  for (int kb = 0; kb < q0blk + 64; kb += 64) {
    // Cooperative stage of K (row-major) and V (transposed).
#pragma unroll
    for (int i = 0; i < 4; i++) {
      int c = tid + i * 128;  // 512 chunks of 8 halves
      int key = c >> 3, d8 = (c & 7) << 3;
      size_t base = ((size_t)(b * T + kb + key)) * 512 + h * 64 + d8;
      *(v8h*)&Ks[key * 72 + d8] = *(const v8h*)(Kg + base);
      v8h vv = *(const v8h*)(Vg + base);
#pragma unroll
      for (int j = 0; j < 8; j++) Vs[(d8 + j) * 72 + key] = vv[j];
    }
    __syncthreads();

    half_t* sw = Ss[wave];
    // Scores for four 16-key sub-tiles, decay applied in registers.
#pragma unroll
    for (int sub = 0; sub < 4; sub++) {
      v8f s = vzero;
      const half_t* krow = &Ks[(sub * 16 + (lane & 15)) * 72 + ((lane < 16) ? 0 : 8)];
#pragma unroll
      for (int st = 0; st < 2; st++) {
        V16U bf;
        bf.h[0] = *(const v8h*)(krow + st * 32);
        bf.h[1] = *(const v8h*)(krow + st * 32 + 16);
        s = wmma_f16(qf[st].v, bf.v, s);
      }
      int keyAbs = kb + sub * 16 + (lane & 15);
      int qrow0 = q0 + ((lane >> 4) << 3);
#pragma unroll
      for (int r = 0; r < 8; r++) {
        int diff = (qrow0 + r) - keyAbs;
        float w = (diff >= 0) ? exp2f(l2g * (float)diff) : 0.f;
        sw[(((lane >> 4) << 3) + r) * 72 + sub * 16 + (lane & 15)] =
            (half_t)(s[r] * 0.125f * w);
      }
    }
    asm volatile("s_wait_dscnt 0" ::: "memory");  // per-wave LDS RAW fence

    // ret += S(16x64) @ V(64x64)
#pragma unroll
    for (int st = 0; st < 2; st++) {
      const half_t* arow = &sw[(lane & 15) * 72 + st * 32 + ((lane < 16) ? 0 : 8)];
      V16U af;
      af.h[0] = *(const v8h*)arow;
      af.h[1] = *(const v8h*)(arow + 16);
#pragma unroll
      for (int dn = 0; dn < 4; dn++) {
        const half_t* vrow =
            &Vs[(dn * 16 + (lane & 15)) * 72 + st * 32 + ((lane < 16) ? 0 : 8)];
        V16U bf;
        bf.h[0] = *(const v8h*)vrow;
        bf.h[1] = *(const v8h*)(vrow + 16);
        racc[dn] = wmma_f16(af.v, bf.v, racc[dn]);
      }
    }
    __syncthreads();
  }

  int qrow0 = q0 + ((lane >> 4) << 3);
#pragma unroll
  for (int dn = 0; dn < 4; dn++)
#pragma unroll
    for (int r = 0; r < 8; r++)
      ret[((size_t)(b * T + qrow0 + r)) * 512 + h * 64 + dn * 16 + (lane & 15)] =
          racc[dn][r];
}

// ---------------------------------------------------------------------------
// Elementwise / normalization kernels
// ---------------------------------------------------------------------------
__global__ void f32_to_f16_kernel(const float* __restrict__ s, half_t* __restrict__ d,
                                  int n) {
  int i = blockIdx.x * 256 + threadIdx.x;
  if (i < n) d[i] = (half_t)s[i];
}

// be/le/re = concat(emb[latent], audio^T) as f16. idx over M*1024.
__global__ void embed_kernel(const int* __restrict__ lat, const float* __restrict__ emb,
                             const float* __restrict__ audio, half_t* __restrict__ out) {
  int idx = blockIdx.x * 256 + threadIdx.x;
  int tok = idx >> 10, c = idx & 1023;
  float v;
  if (c < 512) {
    v = emb[(size_t)lat[tok] * 512 + c];
  } else {
    int b = tok >> 10, t = tok & 1023;
    v = audio[((size_t)b * 512 + (c - 512)) * 1024 + t];
  }
  out[idx] = (half_t)v;
}

// x = tanh(z[:,:512]) * sigmoid(z[:,512:]) where z = whole + class_emb[id].
__global__ void gate_kernel(const float* __restrict__ whole,
                            const float* __restrict__ cemb, const int* __restrict__ id,
                            float* __restrict__ x) {
  int idx = blockIdx.x * 256 + threadIdx.x;  // over M*512
  int tok = idx >> 9, c = idx & 511;
  int b = tok >> 10;
  const float* ce = cemb + (size_t)id[b] * 1024;
  float z1 = whole[(size_t)tok * 1024 + c] + ce[c];
  float z2 = whole[(size_t)tok * 1024 + 512 + c] + ce[512 + c];
  x[(size_t)tok * 512 + c] = tanhf(z1) * (1.f / (1.f + expf(-z2)));
}

// LayerNorm over 512 channels, one token per block (256 threads, 2 ch each).
__global__ __launch_bounds__(256) void layernorm_kernel(const float* __restrict__ x,
                                                        const float* __restrict__ g,
                                                        const float* __restrict__ b,
                                                        half_t* __restrict__ y) {
  __shared__ float redS[8], redQ[8];
  int tok = blockIdx.x;
  int tid = threadIdx.x, lane = tid & 31, wave = tid >> 5;
  const float* row = x + (size_t)tok * 512;
  float v0 = row[tid * 2], v1 = row[tid * 2 + 1];
  float s = v0 + v1, q = v0 * v0 + v1 * v1;
#pragma unroll
  for (int m = 16; m >= 1; m >>= 1) {
    s += __shfl_xor(s, m, 32);
    q += __shfl_xor(q, m, 32);
  }
  if (lane == 0) {
    redS[wave] = s;
    redQ[wave] = q;
  }
  __syncthreads();
  float S = 0.f, Q = 0.f;
#pragma unroll
  for (int i = 0; i < 8; i++) {
    S += redS[i];
    Q += redQ[i];
  }
  float mu = S * (1.f / 512.f);
  float var = Q * (1.f / 512.f) - mu * mu;
  float rs = rsqrtf(var + 1e-5f);
  int c = tid * 2;
  y[(size_t)tok * 512 + c] = (half_t)((v0 - mu) * rs * g[c] + b[c]);
  y[(size_t)tok * 512 + c + 1] = (half_t)((v1 - mu) * rs * g[c + 1] + b[c + 1]);
}

// GroupNorm(8 heads x 64ch) over ret, scale/shift, multiply by silu(gate).
// One token per block; wave w handles head w, lane handles 2 channels.
__global__ __launch_bounds__(256) void gn_gate_kernel(
    const float* __restrict__ ret, const half_t* __restrict__ gsil,
    const float* __restrict__ gn_g, const float* __restrict__ gn_b,
    half_t* __restrict__ msr) {
  int tok = blockIdx.x;
  int wave = threadIdx.x >> 5, lane = threadIdx.x & 31;
  int c0 = wave * 64 + lane * 2;
  const float* row = ret + (size_t)tok * 512;
  float v0 = row[c0], v1 = row[c0 + 1];
  float s = v0 + v1, q = v0 * v0 + v1 * v1;
#pragma unroll
  for (int m = 16; m >= 1; m >>= 1) {
    s += __shfl_xor(s, m, 32);
    q += __shfl_xor(q, m, 32);
  }
  float mu = s * (1.f / 64.f);
  float var = q * (1.f / 64.f) - mu * mu;
  float rs = rsqrtf(var + 1e-5f);
  float r0 = (v0 - mu) * rs * gn_g[c0] + gn_b[c0];
  float r1 = (v1 - mu) * rs * gn_g[c0 + 1] + gn_b[c0 + 1];
  msr[(size_t)tok * 512 + c0] = (half_t)(r0 * (float)gsil[(size_t)tok * 512 + c0]);
  msr[(size_t)tok * 512 + c0 + 1] =
      (half_t)(r1 * (float)gsil[(size_t)tok * 512 + c0 + 1]);
}

// ---------------------------------------------------------------------------
// Host orchestration
// ---------------------------------------------------------------------------
extern "C" void kernel_launch(void* const* d_in, const int* in_sizes, int n_in,
                              void* d_out, int out_size, void* d_ws, size_t ws_size,
                              hipStream_t stream) {
  (void)in_sizes; (void)n_in; (void)out_size; (void)ws_size;
  const int Bc = 8, T = 1024, E = 512, E2 = 1024, E6 = 3072, FFN = 256, VQ = 1024;
  const int M = Bc * T;  // 8192

  // ---- input indices (top-level dict insertion order; params pytree sorted) ----
  const int* lat_body = (const int*)d_in[0];
  const int* lat_lh = (const int*)d_in[1];
  const int* lat_rh = (const int*)d_in[2];
  const float* audio = (const float*)d_in[3];
  const int* idcls = (const int*)d_in[4];
  // params: body_emb=5, body_head{W1=6,W2=7,b=8,b1=9,b2=10,g=11}, class_emb=12,
  // layers[l] base 13+15l {F1,F2,Wg,Wk,Wo,Wq,Wv,fb1,fb2,gn_b,gn_g,ln1_b,ln1_g,ln2_b,ln2_g},
  // lhand_emb=43, lhand_head=44..49, rhand_emb=50, rhand_head=51..56,
  // stack{W1=57,W2=58,W3=59,Wf=60,b1=61,b2=62,b3=63,bf=64}
  const int hIdx[3][6] = {{6, 7, 8, 9, 10, 11},
                          {44, 45, 46, 47, 48, 49},
                          {51, 52, 53, 54, 55, 56}};
  const int embIdx[3] = {5, 43, 50};

  // ---- workspace arena ----
  char* ws = (char*)d_ws;
  size_t cur = 0;
  auto take = [&](size_t bytes) -> char* {
    char* r = ws + cur;
    cur = (cur + bytes + 255) & ~(size_t)255;
    return r;
  };
  half_t* wS1 = (half_t*)take((size_t)E2 * E2 * 2);
  half_t* wS2 = (half_t*)take((size_t)E2 * E2 * 2);
  half_t* wS3 = (half_t*)take((size_t)E2 * E2 * 2);
  half_t* wSf = (half_t*)take((size_t)E6 * E2 * 2);
  half_t *wQ[2], *wK[2], *wV[2], *wG[2], *wO[2], *wF1[2], *wF2[2];
  for (int l = 0; l < 2; l++) {
    wQ[l] = (half_t*)take((size_t)E * E * 2);
    wK[l] = (half_t*)take((size_t)E * E * 2);
    wV[l] = (half_t*)take((size_t)E * E * 2);
    wG[l] = (half_t*)take((size_t)E * E * 2);
    wO[l] = (half_t*)take((size_t)E * E * 2);
    wF1[l] = (half_t*)take((size_t)E * FFN * 2);
    wF2[l] = (half_t*)take((size_t)FFN * E * 2);
  }
  half_t *wH1[3], *wH2[3];
  for (int hd = 0; hd < 3; hd++) {
    wH1[hd] = (half_t*)take((size_t)E * E * 2);
    wH2[hd] = (half_t*)take((size_t)E * VQ * 2);
  }
  char* regionA = take((size_t)3 * M * E2 * 2);  // be/le/re, later q/k/v/g/y
  char* regionB = take((size_t)M * E6 * 2);      // cat3, later ret/msr/ffn/ht
  float* whole = (float*)take((size_t)M * E2 * 4);
  float* xbuf = (float*)take((size_t)M * E * 4);

  half_t* be = (half_t*)regionA;
  half_t* le = be + (size_t)M * E2;
  half_t* re = le + (size_t)M * E2;
  half_t* q16 = (half_t*)regionA;           // alias after gate
  half_t* k16 = q16 + (size_t)M * E;
  half_t* v16w = k16 + (size_t)M * E;
  half_t* g16 = v16w + (size_t)M * E;
  half_t* y16 = g16 + (size_t)M * E;
  half_t* cat3 = (half_t*)regionB;
  float* retb = (float*)regionB;            // alias after whole computed
  half_t* msr16 = (half_t*)(regionB + (size_t)M * E * 4);
  half_t* ffn16 = msr16 + (size_t)M * E;
  half_t* ht16 = ffn16 + (size_t)M * FFN;

  auto conv = [&](int idx, half_t* dst, size_t n) {
    f32_to_f16_kernel<<<dim3((unsigned)((n + 255) / 256)), 256, 0, stream>>>(
        (const float*)d_in[idx], dst, (int)n);
  };

  // ---- weight conversion f32 -> f16 ----
  conv(57, wS1, (size_t)E2 * E2);
  conv(58, wS2, (size_t)E2 * E2);
  conv(59, wS3, (size_t)E2 * E2);
  conv(60, wSf, (size_t)E6 * E2);
  for (int l = 0; l < 2; l++) {
    int pb = 13 + 15 * l;
    conv(pb + 5, wQ[l], (size_t)E * E);
    conv(pb + 3, wK[l], (size_t)E * E);
    conv(pb + 6, wV[l], (size_t)E * E);
    conv(pb + 2, wG[l], (size_t)E * E);
    conv(pb + 4, wO[l], (size_t)E * E);
    conv(pb + 0, wF1[l], (size_t)E * FFN);
    conv(pb + 1, wF2[l], (size_t)FFN * E);
  }
  for (int hd = 0; hd < 3; hd++) {
    conv(hIdx[hd][0], wH1[hd], (size_t)E * E);
    conv(hIdx[hd][1], wH2[hd], (size_t)E * VQ);
  }

  // ---- embeddings + audio concat ----
  dim3 embG((unsigned)((size_t)M * E2 / 256));
  embed_kernel<<<embG, 256, 0, stream>>>(lat_body, (const float*)d_in[embIdx[0]], audio, be);
  embed_kernel<<<embG, 256, 0, stream>>>(lat_lh, (const float*)d_in[embIdx[1]], audio, le);
  embed_kernel<<<embG, 256, 0, stream>>>(lat_rh, (const float*)d_in[embIdx[2]], audio, re);

  // ---- stack: three leaky GEMMs into cat3, then fuse GEMM -> whole ----
  {
    dim3 g(E2 / 64, M / 128);
    gemm_wmma<1, true, false><<<g, 256, 0, stream>>>(
        be, E2, wS1, E2, (const float*)d_in[61], nullptr, 0, cat3 + 0, E6, E2);
    gemm_wmma<1, true, false><<<g, 256, 0, stream>>>(
        le, E2, wS2, E2, (const float*)d_in[62], nullptr, 0, cat3 + E2, E6, E2);
    gemm_wmma<1, true, false><<<g, 256, 0, stream>>>(
        re, E2, wS3, E2, (const float*)d_in[63], nullptr, 0, cat3 + 2 * E2, E6, E2);
    gemm_wmma<0, false, false><<<g, 256, 0, stream>>>(
        cat3, E6, wSf, E2, (const float*)d_in[64], nullptr, 0, whole, E2, E6);
  }

  // ---- gated activation -> x (regionA becomes free for reuse) ----
  gate_kernel<<<dim3((unsigned)((size_t)M * E / 256)), 256, 0, stream>>>(
      whole, (const float*)d_in[12], idcls, xbuf);

  // ---- retnet layers ----
  dim3 gE(E / 64, M / 128);
  for (int l = 0; l < 2; l++) {
    int pb = 13 + 15 * l;
    layernorm_kernel<<<M, 256, 0, stream>>>(xbuf, (const float*)d_in[pb + 12],
                                            (const float*)d_in[pb + 11], y16);
    gemm_wmma<0, true, false><<<gE, 256, 0, stream>>>(y16, E, wQ[l], E, nullptr,
                                                      nullptr, 0, q16, E, E);
    gemm_wmma<0, true, false><<<gE, 256, 0, stream>>>(y16, E, wK[l], E, nullptr,
                                                      nullptr, 0, k16, E, E);
    gemm_wmma<0, true, false><<<gE, 256, 0, stream>>>(y16, E, wV[l], E, nullptr,
                                                      nullptr, 0, v16w, E, E);
    gemm_wmma<3, true, false><<<gE, 256, 0, stream>>>(y16, E, wG[l], E, nullptr,
                                                      nullptr, 0, g16, E, E);
    retention_kernel<<<dim3(T / 64, Bc * 8), 128, 0, stream>>>(q16, k16, v16w, retb);
    gn_gate_kernel<<<M, 256, 0, stream>>>(retb, g16, (const float*)d_in[pb + 10],
                                          (const float*)d_in[pb + 9], msr16);
    gemm_wmma<0, false, true><<<gE, 256, 0, stream>>>(msr16, E, wO[l], E, nullptr,
                                                      xbuf, E, xbuf, E, E);
    layernorm_kernel<<<M, 256, 0, stream>>>(xbuf, (const float*)d_in[pb + 14],
                                            (const float*)d_in[pb + 13], y16);
    gemm_wmma<2, true, false><<<dim3(FFN / 64, M / 128), 256, 0, stream>>>(
        y16, E, wF1[l], FFN, (const float*)d_in[pb + 7], nullptr, 0, ffn16, FFN, E);
    gemm_wmma<0, false, true><<<gE, 256, 0, stream>>>(
        ffn16, FFN, wF2[l], E, (const float*)d_in[pb + 8], xbuf, E, xbuf, E, FFN);
  }

  // ---- output heads ----
  float* outp = (float*)d_out;
  for (int hd = 0; hd < 3; hd++) {
    layernorm_kernel<<<M, 256, 0, stream>>>(xbuf, (const float*)d_in[hIdx[hd][5]],
                                            (const float*)d_in[hIdx[hd][2]], y16);
    gemm_wmma<1, true, false><<<dim3(E / 64, M / 128), 256, 0, stream>>>(
        y16, E, wH1[hd], E, (const float*)d_in[hIdx[hd][3]], nullptr, 0, ht16, E, E);
    gemm_wmma<0, false, false><<<dim3(VQ / 64, M / 128), 256, 0, stream>>>(
        ht16, E, wH2[hd], VQ, (const float*)d_in[hIdx[hd][4]], nullptr, 0,
        outp + (size_t)hd * M * VQ, VQ, E);
  }
}
